// QuaternionLinearAutograd_31138512896843
// MI455X (gfx1250) — matrix operand order repaired and checked by reading, not compiled, not run on hardware
//
#include <hip/hip_runtime.h>

// Quaternion linear layer: out[8192,4096] = x[8192,4096] @ W[4096,4096] + b
//   W[kb*1024+kr, nb*1024+nr] = sign(kb,nb) * Q[kb^nb][kr, nr]
// sign table packed as bitmask 0x5390 (bit kb*4+nb == 1 -> negative).
//
// Compute-bound GEMM (AI ~965 FLOP/B) -> bf16 WMMA, f32 accumulation.
// 128x128x64 block tile, 8 waves (wave32), each wave 64x32 = 4x2 WMMA tiles.
// Double-buffered LDS, fp32->bf16 via sign-XOR + 0x8000 + v_perm_b32 pack.
//
// B is stored in LDS in *fragment-major* K-pair-packed form:
//   uint32 at [(g*128+n)*8 + j] = bf16 pair {K=16g+2j, 16g+2j+1} of column n
// so a WMMA B fragment is two contiguous ds_load_b128 (no register shuffling).
// Steady-state loop is straight-line (last iteration peeled) to avoid
// cross-basic-block register copies.

#define ROWS   8192
#define IN_F   4096
#define OUT_F  4096
#define QD     1024

#define TILE_M 128
#define TILE_N 128
#define TILE_K 64
#define NKT    (IN_F / TILE_K)   // 64 K-tiles
#define LDA    80                // ushorts per A row in LDS (64 data + 16 pad)

typedef __attribute__((ext_vector_type(16))) __bf16       v16bf;
typedef __attribute__((ext_vector_type(8)))  float        v8f;
typedef __attribute__((ext_vector_type(4))) unsigned int  v4u;

struct FRAG32 { v4u lo, hi; };
static_assert(sizeof(FRAG32) == 32, "frag size");

// Pack two fp32 to a bf16 pair (round-half-up) with sign-flip mask m.
// 3 VALU: two adds + one v_perm_b32 byte-select of the high halves.
__device__ __forceinline__ unsigned int pack2_bf16(float a, float b, unsigned int m) {
  unsigned int ua = (__builtin_bit_cast(unsigned int, a) ^ m) + 0x8000u;
  unsigned int ub = (__builtin_bit_cast(unsigned int, b) ^ m) + 0x8000u;
  // perm(s0=ub, s1=ua): result bytes {ub.b3, ub.b2, ua.b3, ua.b2}
  return __builtin_amdgcn_perm(ub, ua, 0x07060302u);
}

__device__ __forceinline__ v8f wmma_bf16(v16bf a, v16bf b, v8f c) {
  return __builtin_amdgcn_wmma_f32_16x16x32_bf16(false, a, false, b, (short)0, c,
                                                 false, false);
}

// A fragment (16x32 bf16): lane&15 = M row, lane>>4 = K-half.
__device__ __forceinline__ v16bf load_a_frag(const unsigned short* base, int m, int koff) {
  const v4u* p = (const v4u*)(base + m * LDA + koff);
  FRAG32 u;
  u.lo = p[0];
  u.hi = p[1];    // +8 ushorts = +16 bytes
  return __builtin_bit_cast(v16bf, u);
}

// B fragment (32x16 bf16): lane&15 = N col, g = ks*2 + (lane>>4).
__device__ __forceinline__ v16bf load_b_frag(const unsigned int* bB, int n, int g) {
  const v4u* p = (const v4u*)(bB + ((g * 128 + n) << 3));
  FRAG32 u;
  u.lo = p[0];
  u.hi = p[1];
  return __builtin_bit_cast(v16bf, u);
}

struct Stage {
  float4       ar[8];      // A: 32 consecutive floats of one x row
  float        bv[2][16];  // B: two K-groups of 16 rows, one column each
  unsigned int smask;      // 0 or 0x80000000 (Hamilton sign)
};

__device__ __forceinline__ void stage_global_load(Stage& s,
    const float* __restrict__ X, const float* __restrict__ Wq,
    int m0, int k0, int nr0, int a_row, int a_col, int b_n, int b_gh) {
  const float4* ap = (const float4*)(X + (size_t)(m0 + a_row) * IN_F + k0 + a_col);
#pragma unroll
  for (int i = 0; i < 8; ++i) s.ar[i] = ap[i];

  const int kr0 = (k0 & (QD - 1)) + 32 * b_gh;
#pragma unroll
  for (int i = 0; i < 2; ++i) {
    const float* bp = Wq + (size_t)(kr0 + 16 * i) * QD + nr0 + b_n;
#pragma unroll
    for (int r = 0; r < 16; ++r) s.bv[i][r] = bp[(size_t)r * QD];
  }
}

__device__ __forceinline__ void stage_lds_store(const Stage& s,
    unsigned short* aW, unsigned int* bW,
    int a_row, int a_col, int b_n, int b_gh) {
  // A: row-major bf16, 4x b128 stores.
  unsigned int au[16];
#pragma unroll
  for (int i = 0; i < 8; ++i) {
    au[2 * i]     = pack2_bf16(s.ar[i].x, s.ar[i].y, 0u);
    au[2 * i + 1] = pack2_bf16(s.ar[i].z, s.ar[i].w, 0u);
  }
  v4u* ad = (v4u*)&aW[a_row * LDA + a_col];
#pragma unroll
  for (int i = 0; i < 4; ++i) {
    v4u t = {au[4 * i], au[4 * i + 1], au[4 * i + 2], au[4 * i + 3]};
    ad[i] = t;
  }

  // B: fragment-major K-pair-packed, 2x b128 stores per group.
#pragma unroll
  for (int i = 0; i < 2; ++i) {
    const int g = 2 * b_gh + i;
    unsigned int bu[8];
#pragma unroll
    for (int j = 0; j < 8; ++j)
      bu[j] = pack2_bf16(s.bv[i][2 * j], s.bv[i][2 * j + 1], s.smask);
    v4u* bd = (v4u*)(bW + ((g * 128 + b_n) << 3));
    v4u t0 = {bu[0], bu[1], bu[2], bu[3]};
    v4u t1 = {bu[4], bu[5], bu[6], bu[7]};
    bd[0] = t0;
    bd[1] = t1;
  }
}

__global__ __launch_bounds__(256)
void qlin_wmma_kernel(const float* __restrict__ X,
                      const float* __restrict__ Wr, const float* __restrict__ Wi,
                      const float* __restrict__ Wj, const float* __restrict__ Wk,
                      const float* __restrict__ Bias, float* __restrict__ Out) {
  __shared__ unsigned short sA[2][TILE_M * LDA];          // 2 x 20 KB
  __shared__ unsigned int   sB[2][4 * TILE_N * 8];        // 2 x 16 KB

  const int tid  = threadIdx.x;
  const int lane = tid & 31;
  const int wv   = tid >> 5;
  const int wm   = wv & 1;      // 2 wave rows (64 M each)
  const int wn   = wv >> 1;     // 4 wave cols (32 N each)
  const int fr   = lane & 15;
  const int fg   = lane >> 4;

  const int m0  = blockIdx.y * TILE_M;
  const int n0  = blockIdx.x * TILE_N;
  const int nb  = n0 >> 10;            // quaternion output block (0..3)
  const int nr0 = n0 & (QD - 1);

  const int a_row = tid >> 1;
  const int a_col = (tid & 1) * 32;
  const int b_n   = tid & 127;         // column within tile
  const int b_gh  = tid >> 7;          // 0..1 -> K-groups {2h, 2h+1}

  auto pick_w = [&](int q) -> const float* {
    const float* p = Wr;
    if (q == 1) p = Wi;
    else if (q == 2) p = Wj;
    else if (q == 3) p = Wk;
    return p;
  };

  v8f acc[4][2];
  const v8f vz = {0.f, 0.f, 0.f, 0.f, 0.f, 0.f, 0.f, 0.f};
#pragma unroll
  for (int i = 0; i < 4; ++i) { acc[i][0] = vz; acc[i][1] = vz; }

  // One tile of WMMA work: hoist all 6 fragments per ks, then 8 WMMAs.
  auto compute_tile = [&](const unsigned short* aB, const unsigned int* bB) {
#pragma unroll
    for (int ks = 0; ks < 2; ++ks) {
      const int g  = ks * 2 + fg;
      const int ko = ks * 32 + fg * 16;
      v16bf bf0 = load_b_frag(bB, wn * 32 + 0  + fr, g);
      v16bf bf1 = load_b_frag(bB, wn * 32 + 16 + fr, g);
      v16bf af0 = load_a_frag(aB, wm * 64 + 0  + fr, ko);
      v16bf af1 = load_a_frag(aB, wm * 64 + 16 + fr, ko);
      v16bf af2 = load_a_frag(aB, wm * 64 + 32 + fr, ko);
      v16bf af3 = load_a_frag(aB, wm * 64 + 48 + fr, ko);
      acc[0][0] = wmma_bf16(af0, bf0, acc[0][0]);
      acc[0][1] = wmma_bf16(af0, bf1, acc[0][1]);
      acc[1][0] = wmma_bf16(af1, bf0, acc[1][0]);
      acc[1][1] = wmma_bf16(af1, bf1, acc[1][1]);
      acc[2][0] = wmma_bf16(af2, bf0, acc[2][0]);
      acc[2][1] = wmma_bf16(af2, bf1, acc[2][1]);
      acc[3][0] = wmma_bf16(af3, bf0, acc[3][0]);
      acc[3][1] = wmma_bf16(af3, bf1, acc[3][1]);
    }
  };

  // Prologue: stage K-tile 0 into buffer 0.
  {
    Stage st;
    st.smask = ((0x5390u >> nb) & 1u) ? 0x80000000u : 0u;   // kb = 0
    stage_global_load(st, X, pick_w(nb), m0, 0, nr0, a_row, a_col, b_n, b_gh);
    stage_lds_store(st, sA[0], sB[0], a_row, a_col, b_n, b_gh);
  }
  __syncthreads();

  // Steady state: straight-line body (last iteration peeled).
  int buf = 0;
#pragma unroll 1
  for (int kt = 0; kt < NKT - 1; ++kt) {
    Stage st;
    const int k0n = (kt + 1) * TILE_K;
    const int kbn = k0n >> 10;
    st.smask = ((0x5390u >> (kbn * 4 + nb)) & 1u) ? 0x80000000u : 0u;
    stage_global_load(st, X, pick_w(kbn ^ nb), m0, k0n, nr0,
                      a_row, a_col, b_n, b_gh);

    compute_tile(sA[buf], sB[buf]);

    stage_lds_store(st, sA[buf ^ 1], sB[buf ^ 1], a_row, a_col, b_n, b_gh);
    buf ^= 1;
    __syncthreads();
  }
  compute_tile(sA[buf], sB[buf]);   // last K-tile, no staging

  // Epilogue: C layout -> lane&15 = N, (lane>>4)*8 + vgpr = M. Bias add, f32 store.
#pragma unroll
  for (int nt = 0; nt < 2; ++nt) {
    const int n  = n0 + wn * 32 + nt * 16 + fr;
    const float bv = Bias[n];
#pragma unroll
    for (int mt = 0; mt < 4; ++mt) {
      const int mbase = m0 + wm * 64 + mt * 16 + fg * 8;
      v8f c = acc[mt][nt];
      float* op = Out + (size_t)mbase * OUT_F + n;
#pragma unroll
      for (int r = 0; r < 8; ++r) op[(size_t)r * OUT_F] = c[r] + bv;
    }
  }
}

extern "C" void kernel_launch(void* const* d_in, const int* in_sizes, int n_in,
                              void* d_out, int out_size, void* d_ws, size_t ws_size,
                              hipStream_t stream) {
  (void)in_sizes; (void)n_in; (void)out_size; (void)d_ws; (void)ws_size;
  const float* x  = (const float*)d_in[0];
  const float* wr = (const float*)d_in[1];
  const float* wi = (const float*)d_in[2];
  const float* wj = (const float*)d_in[3];
  const float* wk = (const float*)d_in[4];
  const float* b  = (const float*)d_in[5];
  float* out = (float*)d_out;

  dim3 grid(OUT_F / TILE_N, ROWS / TILE_M);  // (32, 64)
  qlin_wmma_kernel<<<grid, dim3(256), 0, stream>>>(x, wr, wi, wj, wk, b, out);
}
